// BidirectionalMemory_695784702210
// MI455X (gfx1250) — matrix operand order repaired and checked by reading, not compiled, not run on hardware
//
#include <hip/hip_runtime.h>
#include <stdint.h>

#define BB 4
#define QQ 512
#define MM 8192
#define DD 8
#define EE 512
#define SS 256
#define WST 40   // weight-tile LDS row stride in bf16 elems (80B = 5*16B)
#define BST 36   // sense-tile  LDS row stride in f32  elems (144B = 9*16B)

typedef __attribute__((ext_vector_type(16))) __bf16 v16bf;
typedef __attribute__((ext_vector_type(2)))  __bf16 v2bf;
typedef __attribute__((ext_vector_type(8)))  float  v8f;
typedef __attribute__((ext_vector_type(8)))  unsigned int v8u;

static __device__ __forceinline__ unsigned int f2bf1(float f) {
  union { float f; unsigned u; } v; v.f = f;
  unsigned r = v.u + 0x7FFFu + ((v.u >> 16) & 1u);   // round-to-nearest-even
  return r >> 16;
}
// pack two f32 -> packed bf16x2 (low = first arg).
// Preference: 1x v_cvt_pk_bf16_f32; else 2x v_add + 1x v_perm_b32 (round-half-up);
// else manual RNE.
static __device__ __forceinline__ unsigned int packbf(float lo, float hi) {
#if __has_builtin(__builtin_amdgcn_cvt_pk_bf16_f32)
  v2bf p = __builtin_amdgcn_cvt_pk_bf16_f32(lo, hi);
  return __builtin_bit_cast(unsigned int, p);
#elif __has_builtin(__builtin_amdgcn_perm)
  unsigned a = __builtin_bit_cast(unsigned, lo) + 0x8000u;
  unsigned b = __builtin_bit_cast(unsigned, hi) + 0x8000u;
  return __builtin_amdgcn_perm(b, a, 0x07060302u);   // {b[31:16], a[31:16]}
#else
  return f2bf1(lo) | (f2bf1(hi) << 16);
#endif
}
static __device__ __forceinline__ v16bf make_frag(uint4 lo, uint4 hi) {
  v8u t;
  t[0]=lo.x; t[1]=lo.y; t[2]=lo.z; t[3]=lo.w;
  t[4]=hi.x; t[5]=hi.y; t[6]=hi.z; t[7]=hi.w;
  return __builtin_bit_cast(v16bf, t);
}
// 16 consecutive f32 -> v16bf (works for global or LDS pointers)
static __device__ __forceinline__ v16bf load_cvt16(const float* p) {
  float4 f0 = ((const float4*)p)[0];
  float4 f1 = ((const float4*)p)[1];
  float4 f2 = ((const float4*)p)[2];
  float4 f3 = ((const float4*)p)[3];
  v8u t;
  t[0]=packbf(f0.x,f0.y); t[1]=packbf(f0.z,f0.w);
  t[2]=packbf(f1.x,f1.y); t[3]=packbf(f1.z,f1.w);
  t[4]=packbf(f2.x,f2.y); t[5]=packbf(f2.z,f2.w);
  t[6]=packbf(f3.x,f3.y); t[7]=packbf(f3.z,f3.w);
  return __builtin_bit_cast(v16bf, t);
}

// log-affinity, softmax-invariant constant dropped.
//  - v_rcp_f32 instead of the IEEE divide expansion
//  - sum(log(var_d)) == log(prod(var_d)): one v_log_f32 instead of eight
//    (var in [~0.02, 2] here, so the product stays well inside f32 range)
// Identical in row_stats and gemm1 so exp(l-max)*inv is exact.
static __device__ __forceinline__ float logit8(const float* ql, const float* qs2,
                                               const float* __restrict__ mp,
                                               const float* __restrict__ ms) {
  float acc = 0.f, prod = 1.f;
#pragma unroll
  for (int d = 0; d < DD; ++d) {
    float delta = ql[d] - mp[d];
    float s = ms[d];
    float var = qs2[d] + s * s;
    float tt = (delta * delta) * __builtin_amdgcn_rcpf(var + 1e-8f);
    acc += tt * tt;
    prod *= var;
  }
  return -0.5f * acc - __logf(prod);
}

// ---------------- Kernel 1: per-(b,q) softmax row stats (max, 1/sum) ----------------
__global__ __launch_bounds__(256) void row_stats_kernel(
    const float* __restrict__ loc, const float* __restrict__ lsd,
    const float* __restrict__ mloc, const float* __restrict__ msd,
    float* __restrict__ stats)
{
  const int bq = blockIdx.x;                 // 0 .. B*Q-1
  const int b = bq / QQ, q = bq % QQ;
  const int t = threadIdx.x;

  float ql[DD], qs2[DD];
  {
    const float* lp = loc + ((size_t)b * QQ + q) * DD;
    const float* sp = lsd + ((size_t)b * QQ + q) * DD;
#pragma unroll
    for (int d = 0; d < DD; ++d) { ql[d] = lp[d]; float s = sp[d]; qs2[d] = s * s; }
  }
  const float* mlocB = mloc + (size_t)b * MM * DD;
  const float* msdB  = msd  + (size_t)b * MM * DD;

  float lg[32];
  float lmax = -3.4e38f;
#pragma unroll 4
  for (int i = 0; i < 32; ++i) {
    int m = t + (i << 8);
    float l = logit8(ql, qs2, mlocB + (size_t)m * DD, msdB + (size_t)m * DD);
    lg[i] = l;
    lmax = fmaxf(lmax, l);
  }

  __shared__ float red[256];
  red[t] = lmax; __syncthreads();
  for (int s = 128; s > 0; s >>= 1) {
    if (t < s) red[t] = fmaxf(red[t], red[t + s]);
    __syncthreads();
  }
  const float gmax = red[0];
  __syncthreads();

  float sum = 0.f;
#pragma unroll
  for (int i = 0; i < 32; ++i) sum += __expf(lg[i] - gmax);
  red[t] = sum; __syncthreads();
  for (int s = 128; s > 0; s >>= 1) {
    if (t < s) red[t] += red[t + s];
    __syncthreads();
  }
  if (t == 0) {
    stats[2 * bq]     = gmax;
    stats[2 * bq + 1] = 1.0f / red[0];
  }
}

// ------- Kernel 2: fused softmax-weights x memory_senses -> pre (bf16), WMMA bf16 -------
// Block = 256 thr (8 waves). Tile: 32 q-rows x 128 e-cols, K-loop over M in steps of 32.
// Sense tile is staged f32 straight into LDS (transposed [e][k]) with
// GLOBAL_LOAD_ASYNC_TO_LDS_B32 (ASYNCcnt), converted to bf16 at fragment build.
__global__ __launch_bounds__(256) void gemm1_kernel(
    const float* __restrict__ loc, const float* __restrict__ lsd,
    const float* __restrict__ mloc, const float* __restrict__ msd,
    const float* __restrict__ sense, const float* __restrict__ stats,
    unsigned short* __restrict__ pre)
{
  __shared__ alignas(16) unsigned short smW[32 * WST];  // weights tile [q][k] bf16
  __shared__ alignas(16) float          smBf[128 * BST];// sense tile   [e][k] f32 (transposed)

  const int t  = threadIdx.x;
  const int bx = blockIdx.x;
  const int eblk = bx & 3;
  const int qblk = (bx >> 2) & 15;
  const int b    = bx >> 6;
  const int q0 = qblk * 32;
  const int e0 = eblk * 128;

  // --- logit-producer role: this thread owns query row (q0 + ql_local), 4 k's per step
  const int ql_local = t & 31;
  const int kb = (t >> 5) * 4;
  const int q = q0 + ql_local;
  float ql[DD], qs2[DD];
  {
    const float* lp = loc + ((size_t)b * QQ + q) * DD;
    const float* sp = lsd + ((size_t)b * QQ + q) * DD;
#pragma unroll
    for (int d = 0; d < DD; ++d) { ql[d] = lp[d]; float s = sp[d]; qs2[d] = s * s; }
  }
  const float gmax = stats[2 * (b * QQ + q)];
  const float inv  = stats[2 * (b * QQ + q) + 1];

  // --- async B-stager role: one f32 element per async op, element-granular transpose
  const int e_local = t & 127;
  const int kpar = t >> 7;         // k parity: 0 or 1

  // --- wmma role
  const int lane  = t & 31;
  const int wave  = t >> 5;
  const int qtile = wave >> 2;     // 0..1
  const int eg    = wave & 3;      // 0..3
  const int ln    = lane & 15;
  const int h     = lane >> 4;     // lane group (K-chunk selector per ISA layouts)

  v8f c0 = {}; v8f c1 = {};

  const float* senseB = sense + (size_t)b * MM * EE;
  const float* mlocB  = mloc  + (size_t)b * MM * DD;
  const float* msdB   = msd   + (size_t)b * MM * DD;

  for (int k0 = 0; k0 < MM; k0 += 32) {
    // Phase 1a: async-stage sense tile (32k x 128e) f32 -> LDS [e][k], no VGPR round-trip
#pragma unroll
    for (int j = 0; j < 16; ++j) {
      int k = kpar + 2 * j;        // 0..31, this thread's parity
      unsigned lds_off = (unsigned)(uintptr_t)&smBf[e_local * BST + k];
      unsigned long long ga =
          (unsigned long long)(uintptr_t)(senseB + (size_t)(k0 + k) * EE + e0 + e_local);
      asm volatile("global_load_async_to_lds_b32 %0, %1, off"
                   :: "v"(lds_off), "v"(ga) : "memory");
    }
    // Phase 1b: softmax weight tile (32q x 32k) -> LDS bf16
    {
      float w[4];
#pragma unroll
      for (int j = 0; j < 4; ++j) {
        int m = k0 + kb + j;
        float l = logit8(ql, qs2, mlocB + (size_t)m * DD, msdB + (size_t)m * DD);
        w[j] = __expf(l - gmax) * inv;
      }
      *(uint2*)((char*)smW + ql_local * (WST * 2) + kb * 2) =
          make_uint2(packbf(w[0], w[1]), packbf(w[2], w[3]));
    }
    if (k0 + 32 < MM) {
      __builtin_prefetch(mlocB + (size_t)(k0 + 32 + kb) * DD, 0, 1);
      __builtin_prefetch(msdB  + (size_t)(k0 + 32 + kb) * DD, 0, 1);
    }
    asm volatile("s_wait_asynccnt 0x0" ::: "memory");
    __syncthreads();

    // Phase 2: fragments + WMMA
    const char* wb = (const char*)smW;
    // A fragment (k order shared with B, not the ISA table): lane row = ln, K = h*16 + 0..15 contiguous
    uint4 alo = *(const uint4*)(wb + (qtile * 16 + ln) * (WST * 2) + h * 32);
    uint4 ahi = *(const uint4*)(wb + (qtile * 16 + ln) * (WST * 2) + 16 + h * 32);
    v16bf a = make_frag(alo, ahi);
    // B fragments (k order shared with A, not the ISA table): lane col = ln, K = h*16 + 0..15 contiguous (f32->bf16)
    v16bf bf0 = load_cvt16(&smBf[(eg * 32 + ln)      * BST + h * 16]);
    v16bf bf1 = load_cvt16(&smBf[(eg * 32 + 16 + ln) * BST + h * 16]);

    c0 = __builtin_amdgcn_wmma_f32_16x16x32_bf16(false, a, false, bf0, (short)0, c0, false, false);
    c1 = __builtin_amdgcn_wmma_f32_16x16x32_bf16(false, a, false, bf1, (short)0, c1, false, false);
    __syncthreads();
  }

  // Store pre (bf16, row-major [q][e]) — C layout: VGPR v -> row v+8*h, col ln
  unsigned short* preB = pre + (size_t)b * QQ * EE;
#pragma unroll
  for (int v = 0; v < 8; ++v) {
    int r = q0 + qtile * 16 + v + 8 * h;
    int e = e0 + eg * 32 + ln;
    unsigned p = packbf(c0[v], c1[v]);      // one packed cvt for both columns
    preB[(size_t)r * EE + e]      = (unsigned short)(p & 0xFFFFu);
    preB[(size_t)r * EE + e + 16] = (unsigned short)(p >> 16);
  }
}

// ---------------- Kernel 3: sense = pre(bf16) x W_read^T, WMMA bf16 ----------------
// Block = 256 thr (8 waves) covers 16 q x 256 s; K = E = 512 in steps of 32.
__global__ __launch_bounds__(256) void gemm2_kernel(
    const unsigned short* __restrict__ pre, const float* __restrict__ Wr,
    float* __restrict__ out)
{
  const int t  = threadIdx.x;
  const int bx = blockIdx.x;
  const int qblk = bx & 31;
  const int b    = bx >> 5;
  const int q0 = qblk * 16;
  const int wave = t >> 5;
  const int lane = t & 31;
  const int ln = lane & 15;
  const int h  = lane >> 4;
  const int s0 = wave * 32;

  v8f c0 = {}; v8f c1 = {};
  const unsigned short* preR = pre + ((size_t)b * QQ + q0 + ln) * EE;

#pragma unroll 4
  for (int k0 = 0; k0 < EE; k0 += 32) {
    // A fragment straight from bf16 pre: K = k0 + h*16 + 0..15 contiguous
    uint4 alo = *(const uint4*)(preR + k0 + h * 16);
    uint4 ahi = *(const uint4*)(preR + k0 + 8 + h * 16);
    v16bf a = make_frag(alo, ahi);
    // B fragments: W_read row s, K = k0 + h*16 + 0..15 contiguous f32 -> bf16
    v16bf bf0 = load_cvt16(Wr + (size_t)(s0 + ln)      * EE + k0 + h * 16);
    v16bf bf1 = load_cvt16(Wr + (size_t)(s0 + 16 + ln) * EE + k0 + h * 16);

    c0 = __builtin_amdgcn_wmma_f32_16x16x32_bf16(false, a, false, bf0, (short)0, c0, false, false);
    c1 = __builtin_amdgcn_wmma_f32_16x16x32_bf16(false, a, false, bf1, (short)0, c1, false, false);
  }

  float* outB = out + ((size_t)b * QQ + q0) * SS;
#pragma unroll
  for (int v = 0; v < 8; ++v) {
    int r = v + 8 * h;
    outB[(size_t)r * SS + s0 + ln]      = c0[v];
    outB[(size_t)r * SS + s0 + 16 + ln] = c1[v];
  }
}

extern "C" void kernel_launch(void* const* d_in, const int* in_sizes, int n_in,
                              void* d_out, int out_size, void* d_ws, size_t ws_size,
                              hipStream_t stream) {
  (void)in_sizes; (void)n_in; (void)out_size; (void)ws_size;
  const float* loc   = (const float*)d_in[0];
  const float* lsd   = (const float*)d_in[1];
  const float* mloc  = (const float*)d_in[2];
  const float* msd   = (const float*)d_in[3];
  const float* sense = (const float*)d_in[4];
  const float* Wr    = (const float*)d_in[5];
  float* out = (float*)d_out;

  float* stats = (float*)d_ws;                                      // B*Q*2 f32 = 16 KB
  unsigned short* pre = (unsigned short*)((char*)d_ws + 65536);     // B*Q*E bf16 = 2 MB

  row_stats_kernel<<<BB * QQ, 256, 0, stream>>>(loc, lsd, mloc, msd, stats);
  gemm1_kernel<<<BB * 16 * 4, 256, 0, stream>>>(loc, lsd, mloc, msd, sense, stats, pre);
  gemm2_kernel<<<BB * 32, 256, 0, stream>>>(pre, Wr, out);
}
